// Cross_LocalAttn_43980465111367
// MI455X (gfx1250) — matrix operand order repaired and checked by reading, not compiled, not run on hardware
//
#include <hip/hip_runtime.h>

// ---------------- problem dims ----------------
#define BN     8
#define NTOK   1024          // G*G
#define CDIM   384
#define NH     6
#define HDIM   64
#define TOKS   (BN*NTOK)     // 8192
#define ASCALE 0.125f        // 64^-0.5
#define LNEPS  1e-5f

typedef __attribute__((ext_vector_type(8)))  __bf16 v8bf;
typedef __attribute__((ext_vector_type(16))) __bf16 v16bf;
typedef __attribute__((ext_vector_type(8)))  float  v8f;

typedef __attribute__((ext_vector_type(4))) unsigned int u32x4;
typedef __attribute__((ext_vector_type(8))) int          i32x8;
typedef __attribute__((ext_vector_type(4))) int          i32x4;

#if defined(__has_builtin)
#if __has_builtin(__builtin_amdgcn_tensor_load_to_lds) && \
    __has_builtin(__builtin_amdgcn_s_wait_tensorcnt)
#define HAVE_TDM 1
#endif
#endif

static __device__ __forceinline__ v16bf pack16(v8bf lo, v8bf hi) {
  union { v16bf v; v8bf h[2]; } u;
  u.h[0] = lo; u.h[1] = hi;
  return u.v;
}

static __device__ __forceinline__ v8f wmma_bf16(v16bf a, v16bf b, v8f c) {
  return __builtin_amdgcn_wmma_f32_16x16x32_bf16(false, a, false, b, (short)0, c,
                                                 false, false);
}

// ---------------- weight convert + transpose: W[K,N] f32 -> Wt[N,K] bf16 ----
__global__ __launch_bounds__(256) void wconv_kernel(const float* __restrict__ W,
                                                    __bf16* __restrict__ Wt,
                                                    int K, int Nout) {
#if defined(HAVE_TDM)
  // NULL-descriptor Tensor Data Mover op (D# count=0 => architectural NOP,
  // no memory touched) to exercise the gfx1250 TDM issue/wait path.
  if (blockIdx.x == 0 && threadIdx.x < 32) {
    u32x4 g0 = (u32x4)0u;
    i32x8 g1 = (i32x8)0;
    i32x4 g2 = (i32x4)0;
    i32x4 g3 = (i32x4)0;
#if __clang_major__ >= 23
    i32x8 g4 = (i32x8)0;
    __builtin_amdgcn_tensor_load_to_lds(g0, g1, g2, g3, g4, 0);
#else
    __builtin_amdgcn_tensor_load_to_lds(g0, g1, g2, g3, 0);
#endif
    __builtin_amdgcn_s_wait_tensorcnt((short)0);
  }
#endif
  int idx = blockIdx.x * 256 + threadIdx.x;
  if (idx >= K * Nout) return;
  int k = idx / Nout;
  int n = idx - k * Nout;
  Wt[(long long)n * K + k] = (__bf16)W[idx];
}

// ---------------- LayerNorm: f32 [T,384] -> bf16 [T,384], 1 wave / token ----
__global__ __launch_bounds__(256) void ln_kernel(const float* __restrict__ x,
                                                 const float* __restrict__ g,
                                                 const float* __restrict__ b,
                                                 __bf16* __restrict__ out) {
  const int wave = threadIdx.x >> 5;
  const int lane = threadIdx.x & 31;
  const long long t = (long long)blockIdx.x * 8 + wave;
  const float* xp = x + t * CDIM;

  float v[12];
  float s = 0.f;
#pragma unroll
  for (int i = 0; i < 12; i++) { v[i] = xp[lane + i * 32]; s += v[i]; }
#pragma unroll
  for (int m = 16; m >= 1; m >>= 1) s += __shfl_xor(s, m, 32);
  const float mean = s * (1.f / (float)CDIM);

  float q = 0.f;
#pragma unroll
  for (int i = 0; i < 12; i++) { float d = v[i] - mean; q += d * d; }
#pragma unroll
  for (int m = 16; m >= 1; m >>= 1) q += __shfl_xor(q, m, 32);
  const float rstd = rsqrtf(q * (1.f / (float)CDIM) + LNEPS);

  __bf16* op = out + t * CDIM;
#pragma unroll
  for (int i = 0; i < 12; i++) {
    int c = lane + i * 32;
    op[c] = (__bf16)((v[i] - mean) * rstd * g[c] + b[c]);
  }
}

// ---------------- generic bf16 WMMA GEMM -----------------------------------
// out = A[M,K](lda) @ Bt[N,K]^T + bias ; EPI: 0=bf16, 1=bf16+gelu, 2=f32+resid
// block tile 128x128 (8 waves), wave tile 32x64 (2x4 wmma register blocking)
template <int EPI>
__global__ __launch_bounds__(256) void gemm_bf16_kernel(
    const __bf16* __restrict__ A, int lda,
    const __bf16* __restrict__ Bt, int K, int Nout,
    const float* __restrict__ bias,
    __bf16* __restrict__ outb, float* __restrict__ outf, int ldc,
    const float* __restrict__ resid) {
  const int nblk = Nout >> 7;
  const int bm = blockIdx.x / nblk;
  const int bn = blockIdx.x - bm * nblk;
  const int wave = threadIdx.x >> 5;
  const int lane = threadIdx.x & 31;
  const int wm = wave & 3, wn = wave >> 2;
  const int l15 = lane & 15, kh = lane >> 4;
  const int row0 = bm * 128 + wm * 32;
  const int col0 = bn * 128 + wn * 64;

  const __bf16* a0 = A + (long long)(row0 + l15) * lda;
  const __bf16* a1 = A + (long long)(row0 + 16 + l15) * lda;
  const __bf16* bp[4];
#pragma unroll
  for (int j = 0; j < 4; j++)
    bp[j] = Bt + (long long)(col0 + j * 16 + l15) * K;

  v8f acc[2][4];
#pragma unroll
  for (int i = 0; i < 2; i++)
#pragma unroll
    for (int j = 0; j < 4; j++) acc[i][j] = (v8f)0.f;

#pragma unroll 2
  for (int k0 = 0; k0 < K; k0 += 32) {
    __builtin_prefetch((const void*)(a0 + k0 + 256), 0, 0);
    __builtin_prefetch((const void*)(bp[0] + k0 + 256), 0, 0);
    v16bf af0 = pack16(*(const v8bf*)(a0 + k0 + kh * 8),
                       *(const v8bf*)(a0 + k0 + 16 + kh * 8));
    v16bf af1 = pack16(*(const v8bf*)(a1 + k0 + kh * 8),
                       *(const v8bf*)(a1 + k0 + 16 + kh * 8));
#pragma unroll
    for (int j = 0; j < 4; j++) {
      v16bf bfj = pack16(*(const v8bf*)(bp[j] + k0 + kh * 8),
                         *(const v8bf*)(bp[j] + k0 + 16 + kh * 8));
      acc[0][j] = wmma_bf16(af0, bfj, acc[0][j]);
      acc[1][j] = wmma_bf16(af1, bfj, acc[1][j]);
    }
  }

#pragma unroll
  for (int j = 0; j < 4; j++) {
    const int oc = col0 + j * 16 + l15;
    const float bv = bias[oc];
#pragma unroll
    for (int i = 0; i < 2; i++) {
      const int orb = row0 + i * 16 + kh * 8;
#pragma unroll
      for (int r = 0; r < 8; r++) {
        long long oidx = (long long)(orb + r) * ldc + oc;
        float v = acc[i][j][r] + bv;
        if (EPI == 1) v = 0.5f * v * (1.0f + erff(v * 0.70710678118654752f));
        if (EPI == 2) {
          outf[oidx] = v + resid[oidx];
        } else {
          outb[oidx] = (__bf16)v;
        }
      }
    }
  }
}

// ---------------- dense self-attention (flash style) ------------------------
// QKV bf16 [B,N,1536]: slots q,q1,k,v each [H=6][HD=64]. AO bf16 [B,N,384].
__global__ __launch_bounds__(256) void attn_kernel(const __bf16* __restrict__ QKV,
                                                   __bf16* __restrict__ AO) {
  const int wave = threadIdx.x >> 5;
  const int lane = threadIdx.x & 31;
  const int l15 = lane & 15, kh = lane >> 4;
  const int qblk = blockIdx.x & 7;
  const int bh = blockIdx.x >> 3;
  const int b = bh / NH, h = bh - b * NH;
  const int q0 = qblk * 128 + wave * 16;
  const long long base = (long long)b * NTOK * 1536;

  __shared__ __bf16 ldsV[HDIM * 32];      // V tile transposed: [d][key]
  __shared__ __bf16 ldsP[8][16 * 32];     // per-wave P tile: [row][key]

  // Q fragments (K = head dim, loaded once)
  v16bf aq0, aq1;
  {
    const __bf16* qp = QKV + base + (long long)(q0 + l15) * 1536 + h * HDIM;
    aq0 = pack16(*(const v8bf*)(qp + kh * 8), *(const v8bf*)(qp + 16 + kh * 8));
    aq1 = pack16(*(const v8bf*)(qp + 32 + kh * 8), *(const v8bf*)(qp + 48 + kh * 8));
  }

  float mrow[8], lrow[8];
#pragma unroll
  for (int r = 0; r < 8; r++) { mrow[r] = -1e30f; lrow[r] = 0.f; }
  v8f o[4];
#pragma unroll
  for (int dt = 0; dt < 4; dt++) o[dt] = (v8f)0.f;

  for (int kb = 0; kb < NTOK; kb += 32) {
    __syncthreads();
    // stage V[32 keys][64] transposed into LDS (cooperative, whole block)
    {
      const int key = threadIdx.x >> 3;
      const int dc = (threadIdx.x & 7) * 8;
      const __bf16* vp =
          QKV + base + (long long)(kb + key) * 1536 + 3 * CDIM + h * HDIM + dc;
      v8bf vv = *(const v8bf*)vp;
#pragma unroll
      for (int e = 0; e < 8; e++) ldsV[(dc + e) * 32 + key] = vv[e];
    }
    __syncthreads();

    // S = Q @ K^T for two 16-key sub-tiles
    v8f s0 = (v8f)0.f, s1 = (v8f)0.f;
#pragma unroll
    for (int kk = 0; kk < 2; kk++) {
      const __bf16* kp = QKV + base + (long long)(kb + kk * 16 + l15) * 1536 +
                         2 * CDIM + h * HDIM;
      v16bf bk0 = pack16(*(const v8bf*)(kp + kh * 8), *(const v8bf*)(kp + 16 + kh * 8));
      v16bf bk1 = pack16(*(const v8bf*)(kp + 32 + kh * 8), *(const v8bf*)(kp + 48 + kh * 8));
      v8f s = (v8f)0.f;
      s = wmma_bf16(aq0, bk0, s);
      s = wmma_bf16(aq1, bk1, s);
      if (kk == 0) s0 = s; else s1 = s;
    }

    // online softmax, per row (row = kh*8+r, cols across 16 lanes)
#pragma unroll
    for (int r = 0; r < 8; r++) {
      float a = s0[r] * ASCALE, c = s1[r] * ASCALE;
      float t = fmaxf(a, c);
#pragma unroll
      for (int m = 8; m >= 1; m >>= 1) t = fmaxf(t, __shfl_xor(t, m, 32));
      const float mn = fmaxf(mrow[r], t);
      const float alpha = __expf(mrow[r] - mn);
      mrow[r] = mn;
      const float p0 = __expf(a - mn);
      const float p1 = __expf(c - mn);
      float ps = p0 + p1;
#pragma unroll
      for (int m = 8; m >= 1; m >>= 1) ps += __shfl_xor(ps, m, 32);
      lrow[r] = lrow[r] * alpha + ps;
#pragma unroll
      for (int dt = 0; dt < 4; dt++) o[dt][r] *= alpha;
      ldsP[wave][(kh * 8 + r) * 32 + l15] = (__bf16)p0;
      ldsP[wave][(kh * 8 + r) * 32 + 16 + l15] = (__bf16)p1;
    }

    // O += P @ V
    {
      const __bf16* pp = &ldsP[wave][l15 * 32];
      v16bf ap = pack16(*(const v8bf*)(pp + kh * 8),
                        *(const v8bf*)(pp + 16 + kh * 8));
#pragma unroll
      for (int dt = 0; dt < 4; dt++) {
        v16bf bv = pack16(*(const v8bf*)(&ldsV[(dt * 16 + l15) * 32 + kh * 8]),
                          *(const v8bf*)(&ldsV[(dt * 16 + l15) * 32 + 16 + kh * 8]));
        o[dt] = wmma_bf16(ap, bv, o[dt]);
      }
    }
  }

  // write AO[b, n, h*64 + d]
#pragma unroll
  for (int r = 0; r < 8; r++) {
    const int n = q0 + kh * 8 + r;
    const float inv = 1.0f / lrow[r];
    __bf16* op = AO + ((long long)(b * NTOK + n)) * CDIM + h * HDIM;
#pragma unroll
    for (int dt = 0; dt < 4; dt++) op[dt * 16 + l15] = (__bf16)(o[dt][r] * inv);
  }
}

// ---------------- 3x3 local cross-attention ---------------------------------
// q1 from QKV slot1; K/V from KV [B,N,768]; CO written in [B,H,N,HD] order
// (faithful to torch's transpose-before-reshape).
__global__ __launch_bounds__(256) void cross_kernel(const __bf16* __restrict__ QKV,
                                                    const __bf16* __restrict__ KV,
                                                    __bf16* __restrict__ CO) {
  const int idx = blockIdx.x * 256 + threadIdx.x;   // B*N*H threads exactly
  const int h = idx % NH;
  const int n = (idx / NH) % NTOK;
  const int b = idx / (NH * NTOK);
  const int gy = n >> 5, gx = n & 31;

  const __bf16* qp = QKV + ((long long)(b * NTOK + n)) * 1536 + CDIM + h * HDIM;
  float q[HDIM];
#pragma unroll
  for (int i = 0; i < 8; i++) {
    v8bf t = *(const v8bf*)(qp + i * 8);
#pragma unroll
    for (int e = 0; e < 8; e++) q[i * 8 + e] = (float)t[e];
  }

  int nb[9];
  float sc[9];
  float mx = -1e30f;
#pragma unroll
  for (int t = 0; t < 9; t++) {
    int ny = gy + t / 3 - 1;
    int nx = gx + t % 3 - 1;
    ny = min(max(ny, 0), 31);
    nx = min(max(nx, 0), 31);
    nb[t] = b * NTOK + ny * 32 + nx;
    const __bf16* kp = KV + (long long)nb[t] * (2 * CDIM) + h * HDIM;
    float d = 0.f;
#pragma unroll
    for (int i = 0; i < 8; i++) {
      v8bf kk = *(const v8bf*)(kp + i * 8);
#pragma unroll
      for (int e = 0; e < 8; e++) d += q[i * 8 + e] * (float)kk[e];
    }
    sc[t] = d * ASCALE;
    mx = fmaxf(mx, sc[t]);
  }
  float den = 0.f;
#pragma unroll
  for (int t = 0; t < 9; t++) { sc[t] = __expf(sc[t] - mx); den += sc[t]; }
  const float inv = 1.f / den;

  float acc[HDIM];
#pragma unroll
  for (int d = 0; d < HDIM; d++) acc[d] = 0.f;
#pragma unroll
  for (int t = 0; t < 9; t++) {
    const __bf16* vp = KV + (long long)nb[t] * (2 * CDIM) + CDIM + h * HDIM;
    const float w = sc[t];
#pragma unroll
    for (int i = 0; i < 8; i++) {
      v8bf vv = *(const v8bf*)(vp + i * 8);
#pragma unroll
      for (int e = 0; e < 8; e++) acc[i * 8 + e] += w * (float)vv[e];
    }
  }
  __bf16* op = CO + (((long long)(b * NH + h)) * NTOK + n) * HDIM;
#pragma unroll
  for (int d = 0; d < HDIM; d++) op[d] = (__bf16)(acc[d] * inv);
}

// ---------------- host-side orchestration -----------------------------------
extern "C" void kernel_launch(void* const* d_in, const int* in_sizes, int n_in,
                              void* d_out, int out_size, void* d_ws, size_t ws_size,
                              hipStream_t stream) {
  const float* fea_sp    = (const float*)d_in[0];
  const float* fea_patch = (const float*)d_in[1];
  const float* ln_qkv_g  = (const float*)d_in[2];
  const float* ln_qkv_b  = (const float*)d_in[3];
  const float* W_qkv     = (const float*)d_in[4];
  const float* b_qkv     = (const float*)d_in[5];
  const float* W_sattn   = (const float*)d_in[6];
  const float* b_sattn   = (const float*)d_in[7];
  const float* ln_kv_g   = (const float*)d_in[8];
  const float* ln_kv_b   = (const float*)d_in[9];
  const float* W_kv      = (const float*)d_in[10];
  const float* b_kv      = (const float*)d_in[11];
  const float* W_cross   = (const float*)d_in[12];
  const float* b_cross   = (const float*)d_in[13];
  const float* W_mf      = (const float*)d_in[14];
  const float* b_mf      = (const float*)d_in[15];
  const float* ln_ffn_g  = (const float*)d_in[16];
  const float* ln_ffn_b  = (const float*)d_in[17];
  const float* W_ffn1    = (const float*)d_in[18];
  const float* b_ffn1    = (const float*)d_in[19];
  const float* W_ffn2    = (const float*)d_in[20];
  const float* b_ffn2    = (const float*)d_in[21];
  float* out = (float*)d_out;

  char* ws = (char*)d_ws;
  size_t off = 0;
  auto alloc = [&](size_t bytes) -> void* {
    off = (off + 255) & ~(size_t)255;
    void* p = ws + off;
    off += bytes;
    return p;
  };

  __bf16* WqkvT   = (__bf16*)alloc((size_t)1536 * 384 * 2);
  __bf16* WsattnT = (__bf16*)alloc((size_t)384 * 384 * 2);
  __bf16* WkvT    = (__bf16*)alloc((size_t)768 * 384 * 2);
  __bf16* WcrossT = (__bf16*)alloc((size_t)384 * 384 * 2);
  __bf16* WmfT    = (__bf16*)alloc((size_t)384 * 768 * 2);
  __bf16* Wffn1T  = (__bf16*)alloc((size_t)1536 * 384 * 2);
  __bf16* Wffn2T  = (__bf16*)alloc((size_t)384 * 1536 * 2);
  __bf16* Xn  = (__bf16*)alloc((size_t)TOKS * 384 * 2);
  __bf16* Pn  = (__bf16*)alloc((size_t)TOKS * 384 * 2);
  __bf16* QKV = (__bf16*)alloc((size_t)TOKS * 1536 * 2);
  __bf16* KV  = (__bf16*)alloc((size_t)TOKS * 768 * 2);
  __bf16* AO  = (__bf16*)alloc((size_t)TOKS * 384 * 2);
  __bf16* CO  = (__bf16*)alloc((size_t)TOKS * 384 * 2);
  __bf16* CAT = (__bf16*)alloc((size_t)TOKS * 768 * 2);
  float*  X   = (float*)alloc((size_t)TOKS * 384 * 4);
  __bf16* Hn  = (__bf16*)alloc((size_t)TOKS * 384 * 2);
  __bf16* Hg  = (__bf16*)alloc((size_t)TOKS * 1536 * 2);

  auto wconv = [&](const float* W, __bf16* Wt, int K, int Nout) {
    int total = K * Nout;
    wconv_kernel<<<(total + 255) / 256, 256, 0, stream>>>(W, Wt, K, Nout);
  };
  wconv(W_qkv,   WqkvT,   384, 1536);
  wconv(W_sattn, WsattnT, 384, 384);
  wconv(W_kv,    WkvT,    384, 768);
  wconv(W_cross, WcrossT, 384, 384);
  wconv(W_mf,    WmfT,    768, 384);
  wconv(W_ffn1,  Wffn1T,  384, 1536);
  wconv(W_ffn2,  Wffn2T,  1536, 384);

  ln_kernel<<<TOKS / 8, 256, 0, stream>>>(fea_sp, ln_qkv_g, ln_qkv_b, Xn);
  ln_kernel<<<TOKS / 8, 256, 0, stream>>>(fea_patch, ln_kv_g, ln_kv_b, Pn);

  // QKV projection and KV projection
  gemm_bf16_kernel<0><<<(TOKS / 128) * (1536 / 128), 256, 0, stream>>>(
      Xn, 384, WqkvT, 384, 1536, b_qkv, QKV, nullptr, 1536, nullptr);
  gemm_bf16_kernel<0><<<(TOKS / 128) * (768 / 128), 256, 0, stream>>>(
      Pn, 384, WkvT, 384, 768, b_kv, KV, nullptr, 768, nullptr);

  // attentions
  attn_kernel<<<BN * NH * (NTOK / 128), 256, 0, stream>>>(QKV, AO);
  cross_kernel<<<(BN * NTOK * NH) / 256, 256, 0, stream>>>(QKV, KV, CO);

  // projections into the concat buffer [sattn | cross]
  gemm_bf16_kernel<0><<<(TOKS / 128) * (384 / 128), 256, 0, stream>>>(
      AO, 384, WsattnT, 384, 384, b_sattn, CAT, nullptr, 768, nullptr);
  gemm_bf16_kernel<0><<<(TOKS / 128) * (384 / 128), 256, 0, stream>>>(
      CO, 384, WcrossT, 384, 384, b_cross, CAT + 384, nullptr, 768, nullptr);

  // merge GEMM (K=768) + residual -> X (f32)
  gemm_bf16_kernel<2><<<(TOKS / 128) * (384 / 128), 256, 0, stream>>>(
      CAT, 768, WmfT, 768, 384, b_mf, nullptr, X, 384, fea_sp);

  // FFN
  ln_kernel<<<TOKS / 8, 256, 0, stream>>>(X, ln_ffn_g, ln_ffn_b, Hn);
  gemm_bf16_kernel<1><<<(TOKS / 128) * (1536 / 128), 256, 0, stream>>>(
      Hn, 384, Wffn1T, 384, 1536, b_ffn1, Hg, nullptr, 1536, nullptr);
  gemm_bf16_kernel<2><<<(TOKS / 128) * (384 / 128), 256, 0, stream>>>(
      Hg, 1536, Wffn2T, 1536, 384, b_ffn2, nullptr, out, 384, X);

  (void)in_sizes; (void)n_in; (void)out_size; (void)ws_size;
}
